// VectorQuantize_59820304499392
// MI455X (gfx1250) — matrix operand order repaired and checked by reading, not compiled, not run on hardware
//
#include <hip/hip_runtime.h>

// ---------------------------------------------------------------------------
// VectorQuantize on MI455X (gfx1250): conv1(s2) -> conv2(s2) -> L2-norm ->
// fused (enc @ cbT) + argmax.  GEMMs on v_wmma_f32_16x16x32_bf16; weight /
// codebook / activation tiles staged with gfx1250 async global->LDS DMA
// (ASYNCcnt), double-buffered in the VQ kernel.
// ---------------------------------------------------------------------------

#define USE_ASYNC 1

typedef __attribute__((ext_vector_type(16))) __bf16 v16bf;
typedef __attribute__((ext_vector_type(8)))  float  v8f;

union Frag { unsigned u[8]; v16bf v; };

__device__ __forceinline__ unsigned short f2bf(float f) {
  unsigned u = __float_as_uint(f);
  u += 0x7FFFu + ((u >> 16) & 1u);          // round-to-nearest-even
  return (unsigned short)(u >> 16);
}

// A-fragment K offset for 16-bit A 16x32 (ISA 7.12.2): dword j, lane-half h
__device__ __forceinline__ int koffA(int j, int h) {
  return 2 * j + ((j >> 2) << 3) + (h << 3);
}

// Async global->LDS DMA (GLOBAL_LOAD_ASYNC_TO_LDS_B128, tracked by ASYNCcnt).
// Low 32 bits of a generic pointer to __shared__ == LDS byte offset on AMDGPU.
__device__ __forceinline__ void async_b128(void* lds_dst, const void* gsrc) {
#if USE_ASYNC
  unsigned loff = (unsigned)(size_t)lds_dst;
  unsigned long long ga = (unsigned long long)gsrc;
  asm volatile("global_load_async_to_lds_b128 %0, %1, off"
               :: "v"(loff), "v"(ga) : "memory");
#else
  *(uint4*)lds_dst = *(const uint4*)gsrc;
#endif
}

__device__ __forceinline__ void wait_async0() {
#if USE_ASYNC
#if __has_builtin(__builtin_amdgcn_s_wait_asynccnt)
  __builtin_amdgcn_s_wait_asynccnt(0);
#else
  asm volatile("s_wait_asynccnt 0" ::: "memory");
#endif
#endif
}

// Problem constants
#define T1      2048          // after conv1
#define T2      1024          // after conv2
#define CBN     8192
#define CBD     512

// GEMM macro-tile for the conv kernels
#define TM      64
#define TN      128
#define APITCH  36
#define BPITCH  40            // 80B rows: B128-aligned, 20-bank stride (conflict-free)

// ---------------------------------------------------------------------------
// weight repacks: k-major K ordering so conv K-slices are contiguous in ci,
// and B-tiles become pure bf16 copies (async-DMA-able).
//   wt1[co][kk*1024+ci] = bf16(w1[co][ci][kk])
//   wt2[co][kk* 512+c ] = bf16(w2[co][c ][kk])
// ---------------------------------------------------------------------------
__global__ __launch_bounds__(256) void repack_w1(
    const float* __restrict__ w1, unsigned short* __restrict__ wt1) {
  int i = (int)blockIdx.x * 256 + threadIdx.x;   // 512*3072 total
  if (i >= 512 * 3072) return;
  int co = i / 3072;
  int kg = i - co * 3072;
  int kk = kg >> 10;
  int ci = kg & 1023;
  wt1[i] = f2bf(w1[((size_t)co * 1024 + ci) * 3 + kk]);
}

__global__ __launch_bounds__(256) void repack_w2(
    const float* __restrict__ w2, unsigned short* __restrict__ wt2) {
  int i = (int)blockIdx.x * 256 + threadIdx.x;   // 512*1536 total
  if (i >= 512 * 1536) return;
  int co = i / 1536;
  int kg = i - co * 1536;
  int kk = kg >> 9;
  int c  = kg & 511;
  wt2[i] = f2bf(w2[((size_t)co * 512 + c) * 3 + kk]);
}

// ---------------------------------------------------------------------------
// conv1: x[16,1024,4096] f32 -> y1[(b t1), 512] bf16.  Implicit GEMM:
// M = b*t1 (32768), N = co (512), K = kk*1024+ci (3072).
// Boundary: tin = 2*t1-1+kk in [-1, 4095] -> clamp + select (branch-free).
// ---------------------------------------------------------------------------
__global__ __launch_bounds__(256) void conv1_wmma(
    const float* __restrict__ x, const unsigned short* __restrict__ wt1,
    unsigned short* __restrict__ y1) {
  __shared__ __attribute__((aligned(16))) unsigned short As[TM * APITCH];
  __shared__ __attribute__((aligned(16))) unsigned short Bs[TN * BPITCH];

  const int tid  = threadIdx.x;
  const int lane = tid & 31;
  const int wave = tid >> 5;
  const int h    = (lane >> 4) & 1;
  const int l15  = lane & 15;

  const int mblk = (int)blockIdx.x * TM;     // 512 tiles
  const int nblk = (int)blockIdx.y * TN;     // 4 tiles
  const int m_off  = (wave & 3) * 16;
  const int n_base = (wave >> 2) * 64;

  const int b      = mblk >> 11;             // T1 = 2048
  const int t1base = mblk & (T1 - 1);

  v8f acc[4];
  for (int s = 0; s < 4; ++s)
    for (int i = 0; i < 8; ++i) acc[s][i] = 0.f;

  for (int k0 = 0; k0 < 3072; k0 += 32) {
    // async-stage B: 128 rows x 64B = 512 B128 transfers, 2 per thread
    for (int e = 0; e < 2; ++e) {
      int uidx = e * 256 + tid;
      int n   = uidx >> 2;
      int c16 = uidx & 3;
      async_b128(&Bs[n * BPITCH + c16 * 8],
                 &wt1[(size_t)(nblk + n) * 3072 + k0 + c16 * 8]);
    }
    // sync-stage A (fp32 gather + bf16 convert), dword-packed stores.
    // kk3 is uniform across a 32-wide K-slice (1024 | slice boundaries).
    for (int e = 0; e < 4; ++e) {
      int idx = e * 256 + tid;               // 1024 dwords = 64 rows x 16
      int r = idx >> 4, kp = idx & 15;
      int kg = k0 + kp * 2;
      int kk3 = kg >> 10;
      int ci  = kg & 1023;
      int tin = 2 * (t1base + r) - 1 + kk3;  // in [-1, 4095]
      int tc  = tin < 0 ? 0 : tin;
      const float* xp = &x[((size_t)b * 1024 + ci) * 4096 + tc];
      float v0 = xp[0];
      float v1 = xp[4096];
      if (tin < 0) { v0 = 0.f; v1 = 0.f; }
      *(unsigned*)&As[r * APITCH + kp * 2] =
          (unsigned)f2bf(v0) | ((unsigned)f2bf(v1) << 16);
    }
    wait_async0();
    __syncthreads();

    Frag a;
    const int arow = m_off + l15;
    for (int j = 0; j < 8; ++j)
      a.u[j] = *(const unsigned*)&As[arow * APITCH + koffA(j, h)];

    for (int s = 0; s < 4; ++s) {
      Frag bf;
      const int brow = n_base + s * 16 + l15;
      for (int j = 0; j < 8; ++j)
        bf.u[j] = *(const unsigned*)&Bs[brow * BPITCH + koffA(j, h)];
      acc[s] = __builtin_amdgcn_wmma_f32_16x16x32_bf16(
          false, a.v, false, bf.v, (short)0, acc[s], false, false);
    }
    __syncthreads();
  }

  for (int s = 0; s < 4; ++s)
    for (int r = 0; r < 8; ++r) {
      int gm = mblk + m_off + h * 8 + r;
      int gn = nblk + n_base + s * 16 + l15;
      y1[(size_t)gm * 512 + gn] = f2bf(acc[s][r]);
    }
}

// ---------------------------------------------------------------------------
// conv2: y1 bf16 [(b t1), 512] -> xe f32 [(b t2), 512].
// M = b*t2 (16384), N = co (512), K = kk*512 + c (1536).
// ---------------------------------------------------------------------------
__global__ __launch_bounds__(256) void conv2_wmma(
    const unsigned short* __restrict__ y1, const unsigned short* __restrict__ wt2,
    float* __restrict__ xe) {
  __shared__ __attribute__((aligned(16))) unsigned short As[TM * APITCH];
  __shared__ __attribute__((aligned(16))) unsigned short Bs[TN * BPITCH];

  const int tid  = threadIdx.x;
  const int lane = tid & 31;
  const int wave = tid >> 5;
  const int h    = (lane >> 4) & 1;
  const int l15  = lane & 15;

  const int mblk = (int)blockIdx.x * TM;     // 256 tiles
  const int nblk = (int)blockIdx.y * TN;     // 4 tiles
  const int m_off  = (wave & 3) * 16;
  const int n_base = (wave >> 2) * 64;

  const int b      = mblk >> 10;             // T2 = 1024
  const int t2base = mblk & (T2 - 1);

  v8f acc[4];
  for (int s = 0; s < 4; ++s)
    for (int i = 0; i < 8; ++i) acc[s][i] = 0.f;

  for (int k0 = 0; k0 < 1536; k0 += 32) {
    // async-stage B: 512 B128, 2 per thread
    for (int e = 0; e < 2; ++e) {
      int uidx = e * 256 + tid;
      int n   = uidx >> 2;
      int c16 = uidx & 3;
      async_b128(&Bs[n * BPITCH + c16 * 8],
                 &wt2[(size_t)(nblk + n) * 1536 + k0 + c16 * 8]);
    }
    // sync-stage A: bf16 rows, dword copies; tin in [-1, 2047] -> clamp+select
    for (int e = 0; e < 4; ++e) {
      int idx = e * 256 + tid;               // 1024 dwords
      int r = idx >> 4, kp = idx & 15;
      int kg = k0 + kp * 2;
      int kk3 = kg >> 9;
      int c   = kg & 511;
      int tin = 2 * (t2base + r) - 1 + kk3;
      int tc  = tin < 0 ? 0 : tin;
      unsigned v = *(const unsigned*)&y1[((size_t)b * T1 + tc) * 512 + c];
      if (tin < 0) v = 0u;
      *(unsigned*)&As[r * APITCH + kp * 2] = v;
    }
    wait_async0();
    __syncthreads();

    Frag a;
    const int arow = m_off + l15;
    for (int j = 0; j < 8; ++j)
      a.u[j] = *(const unsigned*)&As[arow * APITCH + koffA(j, h)];

    for (int s = 0; s < 4; ++s) {
      Frag bf;
      const int brow = n_base + s * 16 + l15;
      for (int j = 0; j < 8; ++j)
        bf.u[j] = *(const unsigned*)&Bs[brow * BPITCH + koffA(j, h)];
      acc[s] = __builtin_amdgcn_wmma_f32_16x16x32_bf16(
          false, a.v, false, bf.v, (short)0, acc[s], false, false);
    }
    __syncthreads();
  }

  for (int s = 0; s < 4; ++s)
    for (int r = 0; r < 8; ++r) {
      int gm = mblk + m_off + h * 8 + r;
      int gn = nblk + n_base + s * 16 + l15;
      xe[(size_t)gm * 512 + gn] = acc[s][r];
    }
}

// ---------------------------------------------------------------------------
// Row-wise L2 normalize (f32 in) -> bf16 out.  One block per row.
// ---------------------------------------------------------------------------
__global__ __launch_bounds__(256) void l2norm_bf16(
    const float* __restrict__ in, unsigned short* __restrict__ out, int cols) {
  __shared__ float sred[256];
  const int row = (int)blockIdx.x;
  const float* rp = in + (size_t)row * cols;
  float ss = 0.f;
  for (int c = threadIdx.x; c < cols; c += 256) { float v = rp[c]; ss += v * v; }
  sred[threadIdx.x] = ss;
  __syncthreads();
  for (int off = 128; off > 0; off >>= 1) {
    if ((int)threadIdx.x < off) sred[threadIdx.x] += sred[threadIdx.x + off];
    __syncthreads();
  }
  const float scale = 1.f / fmaxf(sqrtf(sred[0]), 1e-12f);
  for (int c = threadIdx.x; c < cols; c += 256)
    out[(size_t)row * cols + c] = f2bf(rp[c] * scale);
}

// ---------------------------------------------------------------------------
// Fused VQ: argmax_n ( enc_n[m,:] . cbn[n,:] )  (== argmin distance).
// A-tile: 32 rows x full K=512, async-staged once.  B-tiles: double-buffered
// async pipeline, one barrier per K-step, pipelined across codebook chunks.
// ---------------------------------------------------------------------------
#define QM 32
#define QAPITCH 520
__global__ __launch_bounds__(256) void vq_argmax(
    const unsigned short* __restrict__ enc, const unsigned short* __restrict__ cbn,
    float* __restrict__ out) {
  __shared__ __attribute__((aligned(16))) unsigned short As[QM * QAPITCH];
  __shared__ __attribute__((aligned(16))) unsigned short Bs[2][128 * BPITCH];
  __shared__ float cval[QM * 4];
  __shared__ int   cidx[QM * 4];

  const int tid  = threadIdx.x;
  const int lane = tid & 31;
  const int wave = tid >> 5;
  const int h    = (lane >> 4) & 1;
  const int l15  = lane & 15;

  const int mblk  = (int)blockIdx.x * QM;   // 512 blocks
  const int m_off = (wave & 1) * 16;
  const int nq    = wave >> 1;              // 0..3

  // async-stage full A tile once: 32 rows x 1024B = 2048 B128, 8 per thread
  for (int e = 0; e < 8; ++e) {
    int uidx = e * 256 + tid;
    int r  = uidx >> 6;                     // 64 B128 per row
    int cu = uidx & 63;
    async_b128(&As[r * QAPITCH + cu * 8],
               &enc[(size_t)(mblk + r) * 512 + cu * 8]);
  }

  // B staging: 128 rows x 64B = 512 B128, 2 per thread
  auto stageB = [&](int buf, int nc, int k0) {
    for (int e = 0; e < 2; ++e) {
      int uidx = e * 256 + tid;
      int n   = uidx >> 2;
      int c16 = uidx & 3;
      async_b128(&Bs[buf][n * BPITCH + c16 * 8],
                 &cbn[(size_t)(nc + n) * 512 + k0 + c16 * 8]);
    }
  };

  float bestv[8];
  int   besti[8];
  for (int r = 0; r < 8; ++r) { bestv[r] = -3.402823466e+38f; besti[r] = 0; }

  const int NSTEP = (CBN / 128) * 16;       // 64 chunks x 16 K-steps
  stageB(0, 0, 0);

  v8f acc[2];
  for (int s = 0; s < NSTEP; ++s) {
    wait_async0();
    __syncthreads();                        // buf[s&1] ready; prev reads done
    if (s + 1 < NSTEP)
      stageB((s + 1) & 1, ((s + 1) >> 4) * 128, ((s + 1) & 15) * 32);

    const int k0 = (s & 15) * 32;
    if ((s & 15) == 0)
      for (int t = 0; t < 2; ++t)
        for (int i = 0; i < 8; ++i) acc[t][i] = 0.f;

    Frag a;
    const int arow = m_off + l15;
    for (int j = 0; j < 8; ++j)
      a.u[j] = *(const unsigned*)&As[arow * QAPITCH + k0 + koffA(j, h)];

    for (int t = 0; t < 2; ++t) {
      Frag bf;
      const int brow = nq * 32 + t * 16 + l15;
      for (int j = 0; j < 8; ++j)
        bf.u[j] = *(const unsigned*)&Bs[s & 1][brow * BPITCH + k0 * 0 + koffA(j, h)];
      acc[t] = __builtin_amdgcn_wmma_f32_16x16x32_bf16(
          false, a.v, false, bf.v, (short)0, acc[t], false, false);
    }

    if ((s & 15) == 15) {                   // end of codebook chunk: fold
      const int nc = (s >> 4) * 128;
      for (int t = 0; t < 2; ++t)
        for (int r = 0; r < 8; ++r) {
          float v = acc[t][r];
          int n = nc + nq * 32 + t * 16 + l15;
          if (v > bestv[r]) { bestv[r] = v; besti[r] = n; }   // ascending n
        }
    }
  }

  // butterfly reduce across the 16 lanes of each half-wave (wave32 shuffles)
  for (int mask = 1; mask <= 8; mask <<= 1)
    for (int r = 0; r < 8; ++r) {
      float ov = __shfl_xor(bestv[r], mask, 32);
      int   oi = __shfl_xor(besti[r], mask, 32);
      if (ov > bestv[r] || (ov == bestv[r] && oi < besti[r])) {
        bestv[r] = ov; besti[r] = oi;
      }
    }
  if (l15 == 0) {
    for (int r = 0; r < 8; ++r) {
      int m = m_off + h * 8 + r;            // 0..31
      cval[m * 4 + nq] = bestv[r];
      cidx[m * 4 + nq] = besti[r];
    }
  }
  __syncthreads();

  if (tid < QM) {
    float bv = -3.402823466e+38f;
    int bi = 0x7fffffff;
    for (int q = 0; q < 4; ++q) {
      float v = cval[tid * 4 + q];
      int  i2 = cidx[tid * 4 + q];
      if (v > bv || (v == bv && i2 < bi)) { bv = v; bi = i2; }
    }
    out[mblk + tid] = (float)bi;            // output dtype: float
  }
}

// ---------------------------------------------------------------------------
// Host-side orchestration
// ---------------------------------------------------------------------------
extern "C" void kernel_launch(void* const* d_in, const int* in_sizes, int n_in,
                              void* d_out, int out_size, void* d_ws, size_t ws_size,
                              hipStream_t stream) {
  const float* x  = (const float*)d_in[0];   // [16,1024,4096]
  const float* w1 = (const float*)d_in[1];   // [512,1024,3]
  const float* w2 = (const float*)d_in[2];   // [512,512,3]
  const float* cb = (const float*)d_in[3];   // [8192,512]
  float* out = (float*)d_out;                // 16384 indices (as float)

  char* ws = (char*)d_ws;
  unsigned short* y1    = (unsigned short*)(ws);              // 33,554,432 B
  float*          xe    = (float*)(ws + 33554432);            // 33,554,432 B
  unsigned short* enc_n = (unsigned short*)(ws + 67108864);   // 16,777,216 B
  unsigned short* cbn   = (unsigned short*)(ws + 83886080);   //  8,388,608 B
  unsigned short* wt2   = (unsigned short*)(ws + 92274688);   //  1,572,864 B
  unsigned short* wt1   = (unsigned short*)(ws + 93847552);   //  3,145,728 B

  repack_w1<<<dim3(6144), 256, 0, stream>>>(w1, wt1);
  repack_w2<<<dim3(3072), 256, 0, stream>>>(w2, wt2);
  conv1_wmma<<<dim3(512, 4), 256, 0, stream>>>(x, wt1, y1);
  conv2_wmma<<<dim3(256, 4), 256, 0, stream>>>(y1, wt2, xe);
  l2norm_bf16<<<dim3(16384), 256, 0, stream>>>(xe, enc_n, 512);
  l2norm_bf16<<<dim3(8192), 256, 0, stream>>>(cb, cbn, 512);
  vq_argmax<<<dim3(512), 256, 0, stream>>>(enc_n, cbn, out);
}
